// MultiHeadSelfAttention_46033459478588
// MI455X (gfx1250) — matrix operand order repaired and checked by reading, not compiled, not run on hardware
//
#include <hip/hip_runtime.h>
#include <stdint.h>

// ---------------------------------------------------------------------------
// MHA with RoPE for MI455X (gfx1250, wave32, WMMA bf16 -> f32 accum)
// b=2, s=2048, d=1024, h=16, d_k=64
// ---------------------------------------------------------------------------

#define B_   2
#define S_   2048
#define D_   1024
#define H_   16
#define DK_  64
#define NTOK (B_ * S_)          // 4096

typedef __attribute__((ext_vector_type(16))) __bf16 v16bf;
typedef __attribute__((ext_vector_type(8)))  float  v8f;

union V16U { v16bf v; uint4 q[2]; };
union V8FU { v8f v; float f[8]; };

// ---- bf16 helpers (RNE) ----------------------------------------------------
__device__ __forceinline__ uint16_t f2bf(float x) {
    uint32_t u = __builtin_bit_cast(uint32_t, x);
    u = (u + 0x7FFFu + ((u >> 16) & 1u)) >> 16;
    return (uint16_t)u;
}
__device__ __forceinline__ float bf2f(uint16_t h) {
    uint32_t u = ((uint32_t)h) << 16;
    return __builtin_bit_cast(float, u);
}
__device__ __forceinline__ uint32_t pack2(float a, float b) {
    return (uint32_t)f2bf(a) | ((uint32_t)f2bf(b) << 16);
}
// exchange with lane ^ 16 (SWAPX16: xor=0x10, and=0x1f -> offset 0x401F)
__device__ __forceinline__ float swap16f(float x) {
    int i = __builtin_bit_cast(int, x);
    i = __builtin_amdgcn_ds_swizzle(i, 0x401F);
    return __builtin_bit_cast(float, i);
}

// ---- WMMA operand loaders (bf16, K-chunk of 32) ----------------------------
// A operand as loaded here (contiguous k, same as B): lane row = M, VGPRs 0..7: K = half*16 + 2v,2v+1
__device__ __forceinline__ v16bf load_a16(const uint16_t* row, int d0, int half) {
    V16U u;
    u.q[0] = *(const uint4*)(row + d0 + 16 * half);
    u.q[1] = *(const uint4*)(row + d0 + 16 * half + 8);
    return u.v;
}
// B operand as loaded here (contiguous k, same as A): lane col = N, VGPRs 0..7: K = half*16 + 2v,2v+1
__device__ __forceinline__ v16bf load_b16(const uint16_t* row, int d0, int half) {
    V16U u;
    const uint4* p = (const uint4*)(row + d0 + 16 * half);
    u.q[0] = p[0];
    u.q[1] = p[1];
    return u.v;
}

#define WMMA_BF16(a, b, c) \
    __builtin_amdgcn_wmma_f32_16x16x32_bf16(false, (a), false, (b), (short)0, (c), false, false)

// ---------------------------------------------------------------------------
// f32 -> bf16 convert
// ---------------------------------------------------------------------------
__global__ __launch_bounds__(256) void cvt_f32_bf16(const float* __restrict__ in,
                                                    uint16_t* __restrict__ out, int n) {
    int i = blockIdx.x * 256 + threadIdx.x;
    if (i < n) out[i] = f2bf(in[i]);
}

// ---------------------------------------------------------------------------
// GEMM: Y[n,i] = sum_j W[i,j] * X[n,j]     (X: [NTOK][K] bf16, W: [Dout][K] bf16)
// Wave tile: 32 features (2 x M=16) x 64 tokens (4 x N=16). Y token-major.
// ---------------------------------------------------------------------------
__global__ __launch_bounds__(256) void gemm_wt_xt(const uint16_t* __restrict__ X,
                                                  const uint16_t* __restrict__ W,
                                                  void* __restrict__ Y,
                                                  int K, int Dout, int fTiles,
                                                  int outF32) {
    const int wave   = blockIdx.x * 8 + (threadIdx.x >> 5);
    const int lane   = threadIdx.x & 31;
    const int lane16 = lane & 15;
    const int half   = lane >> 4;

    const int fT = wave % fTiles;          // fTiles = Dout/32
    const int tT = wave / fTiles;
    const int feat0 = fT * 32;
    const int tok0  = tT * 64;

    const uint16_t* arow0 = W + (size_t)(feat0 + lane16) * K;
    const uint16_t* arow1 = W + (size_t)(feat0 + 16 + lane16) * K;
    const uint16_t* brow0 = X + (size_t)(tok0 + 0  + lane16) * K;
    const uint16_t* brow1 = X + (size_t)(tok0 + 16 + lane16) * K;
    const uint16_t* brow2 = X + (size_t)(tok0 + 32 + lane16) * K;
    const uint16_t* brow3 = X + (size_t)(tok0 + 48 + lane16) * K;

    V8FU acc[2][4];
#pragma unroll
    for (int f = 0; f < 2; f++)
#pragma unroll
        for (int s = 0; s < 4; s++)
#pragma unroll
            for (int r = 0; r < 8; r++) acc[f][s].f[r] = 0.f;

    for (int kk = 0; kk < K; kk += 32) {
        v16bf a0 = load_a16(arow0, kk, half);
        v16bf a1 = load_a16(arow1, kk, half);
        v16bf b0 = load_b16(brow0, kk, half);
        v16bf b1 = load_b16(brow1, kk, half);
        v16bf b2 = load_b16(brow2, kk, half);
        v16bf b3 = load_b16(brow3, kk, half);
        acc[0][0].v = WMMA_BF16(a0, b0, acc[0][0].v);
        acc[0][1].v = WMMA_BF16(a0, b1, acc[0][1].v);
        acc[0][2].v = WMMA_BF16(a0, b2, acc[0][2].v);
        acc[0][3].v = WMMA_BF16(a0, b3, acc[0][3].v);
        acc[1][0].v = WMMA_BF16(a1, b0, acc[1][0].v);
        acc[1][1].v = WMMA_BF16(a1, b1, acc[1][1].v);
        acc[1][2].v = WMMA_BF16(a1, b2, acc[1][2].v);
        acc[1][3].v = WMMA_BF16(a1, b3, acc[1][3].v);
    }

#pragma unroll
    for (int f = 0; f < 2; f++) {
        const int fOff = feat0 + 16 * f + 8 * half;   // 8 contiguous features/lane
#pragma unroll
        for (int s = 0; s < 4; s++) {
            V8FU* a = &acc[f][s];
            const size_t tok = (size_t)(tok0 + 16 * s + lane16);
            if (outF32) {
                float* y = (float*)Y + tok * Dout + fOff;
                *(float4*)(y)     = make_float4(a->f[0], a->f[1], a->f[2], a->f[3]);
                *(float4*)(y + 4) = make_float4(a->f[4], a->f[5], a->f[6], a->f[7]);
            } else {
                uint16_t* y = (uint16_t*)Y + tok * Dout + fOff;
                uint4 w;
                w.x = pack2(a->f[0], a->f[1]);
                w.y = pack2(a->f[2], a->f[3]);
                w.z = pack2(a->f[4], a->f[5]);
                w.w = pack2(a->f[6], a->f[7]);
                *(uint4*)y = w;
            }
        }
    }
}

// ---------------------------------------------------------------------------
// RoPE in place on bf16 buffer [NTOK][1024]; pairs (2f, 2f+1) within each head
// ---------------------------------------------------------------------------
__global__ __launch_bounds__(256) void rope_kernel(uint16_t* __restrict__ buf,
                                                   const int* __restrict__ tokpos) {
    int tid = blockIdx.x * 256 + threadIdx.x;           // NTOK * 512 threads
    int tok = tid >> 9;
    int pp  = tid & 511;
    int h   = pp >> 5;
    int f   = pp & 31;
    float pos = (float)tokpos[tok & (S_ - 1)];
    float freq = __expf(-(float)f * (9.210340371976184f / 32.0f)); // theta^(-2f/64)
    float ang = pos * freq;
    float c, s;
    __sincosf(ang, &s, &c);
    uint32_t* p = (uint32_t*)(buf + (size_t)tok * D_ + h * DK_ + 2 * f);
    uint32_t w = *p;
    float x1 = bf2f((uint16_t)(w & 0xFFFF));
    float x2 = bf2f((uint16_t)(w >> 16));
    *p = pack2(x1 * c - x2 * s, x2 * c + x1 * s);
}

// ---------------------------------------------------------------------------
// V transpose: Vt[b][h][dv][tok] = V[b*S+tok][h*64+dv]
// ---------------------------------------------------------------------------
__global__ __launch_bounds__(256) void transpose_v(const uint16_t* __restrict__ V,
                                                   uint16_t* __restrict__ Vt) {
    int tid = blockIdx.x * 256 + threadIdx.x;            // B*H*DK*S threads
    int tok = tid & (S_ - 1);
    int r   = tid >> 11;
    int dv  = r & (DK_ - 1);
    int r2  = r >> 6;
    int h   = r2 & (H_ - 1);
    int b   = r2 >> 4;
    Vt[tid] = V[((size_t)(b * S_ + tok)) * D_ + h * DK_ + dv];
}

// ---------------------------------------------------------------------------
// Flash attention. One block (8 waves) = one (b,h) and 128 consecutive queries;
// waves own 16-query tiles. K/Vt 32-k chunks are staged in LDS once per block
// (8x cut in L2 traffic). S^T = K x Q^T so softmax rows are lane-local + one
// SWAPX16; O^T += Vt x P^T with P repacked into the B layout via SWAPX16.
// ---------------------------------------------------------------------------
__global__ __launch_bounds__(256) void attn_kernel(const uint16_t* __restrict__ Q,
                                                   const uint16_t* __restrict__ Kb,
                                                   const uint16_t* __restrict__ Vt,
                                                   uint16_t* __restrict__ O) {
    __shared__ uint16_t kT[32 * DK_];   // 32 k-rows x 64 d        (4 KB)
    __shared__ uint16_t vT[DK_ * 32];   // 64 dv-rows x 32 k       (4 KB)

    const int tid    = threadIdx.x;
    const int wid    = tid >> 5;
    const int lane   = tid & 31;
    const int lane16 = lane & 15;
    const int half   = lane >> 4;

    const int blk  = blockIdx.x;        // B*H*(S/128) = 512 blocks
    const int b    = blk >> 8;
    const int h    = (blk >> 4) & (H_ - 1);
    const int qblk = blk & 15;

    const int q0  = qblk * 128 + wid * 16;
    const int myq = q0 + lane16;

    const size_t tokBase = (size_t)b * S_;

    // Q operand (B matrix), loaded once, reused for the whole k loop
    const uint16_t* qrow = Q + (tokBase + q0 + lane16) * D_ + h * DK_;
    const v16bf bq0 = load_b16(qrow, 0, half);
    const v16bf bq1 = load_b16(qrow, 32, half);

    // LDS fill addressing (256 threads -> one 16B segment each per tile)
    const int kRow = tid >> 3, kSeg = tid & 7;   // K tile: 32 rows x 8 segs
    const int vRow = tid >> 2, vSeg = tid & 3;   // V tile: 64 rows x 4 segs
    const uint16_t* kSrcBase = Kb + (tokBase + kRow) * D_ + h * DK_ + kSeg * 8;
    const uint16_t* vSrcBase = Vt + ((size_t)(b * H_ + h) * DK_ + vRow) * S_ + vSeg * 8;
    uint16_t* kDst = kT + kRow * DK_ + kSeg * 8;
    uint16_t* vDst = vT + vRow * 32 + vSeg * 8;

    V8FU acc[4];
#pragma unroll
    for (int t = 0; t < 4; t++)
#pragma unroll
        for (int r = 0; r < 8; r++) acc[t].f[r] = 0.f;

    float m = -3.0e38f;
    float l = 0.f;
    const float scale = 0.125f;  // 1/sqrt(64)

    const int kmaxBlk = qblk * 128 + 127;     // block-uniform loop bound
    const int kendW   = q0 + 15;              // wave-uniform compute bound

    for (int k0 = 0; k0 <= kmaxBlk; k0 += 32) {
        __syncthreads();                       // previous chunk fully consumed
        *(uint4*)kDst = *(const uint4*)(kSrcBase + (size_t)k0 * D_);
        *(uint4*)vDst = *(const uint4*)(vSrcBase + k0);
        if (k0 + 32 <= kmaxBlk) {              // prefetch next chunk
            __builtin_prefetch(kSrcBase + (size_t)(k0 + 32) * D_, 0, 1);
            __builtin_prefetch(vSrcBase + (k0 + 32), 0, 1);
        }
        __syncthreads();                       // chunk visible to all waves

        if (k0 > kendW) continue;              // wave-uniform: EXEC stays full

        const uint16_t* ka0 = kT;              // rows 0..15
        const uint16_t* ka1 = kT + 16 * DK_;   // rows 16..31

        V8FU s0, s1;
        {
            v8f z0 = {};
            z0 = WMMA_BF16(load_a16(ka0 + lane16 * DK_ - lane16 * DK_ + (size_t)lane16 * DK_, 0, half), bq0, z0);
            z0 = WMMA_BF16(load_a16(ka0 + (size_t)lane16 * DK_, 32, half), bq1, z0);
            s0.v = z0;
            v8f z1 = {};
            z1 = WMMA_BF16(load_a16(ka1 + (size_t)lane16 * DK_, 0, half), bq0, z1);
            z1 = WMMA_BF16(load_a16(ka1 + (size_t)lane16 * DK_, 32, half), bq1, z1);
            s1.v = z1;
        }

        // scale, causal mask, chunk max
        float cmax = -3.0e38f;
#pragma unroll
        for (int r = 0; r < 8; r++) {
            int kr0 = k0 + 8 * half + r;
            int kr1 = kr0 + 16;
            float v0 = s0.f[r] * scale; if (kr0 > myq) v0 = -3.0e38f;
            float v1 = s1.f[r] * scale; if (kr1 > myq) v1 = -3.0e38f;
            s0.f[r] = v0; s1.f[r] = v1;
            cmax = fmaxf(cmax, fmaxf(v0, v1));
        }
        cmax = fmaxf(cmax, swap16f(cmax));          // full 32-k row max
        const float mn = fmaxf(m, cmax);
        const float alpha = __expf(m - mn);
        m = mn;

        float p0[8], p1[8];
        float ls = 0.f;
#pragma unroll
        for (int r = 0; r < 8; r++) {
            p0[r] = __expf(s0.f[r] - mn);
            p1[r] = __expf(s1.f[r] - mn);
            ls += p0[r] + p1[r];
        }
        ls += swap16f(ls);
        l = l * alpha + ls;

#pragma unroll
        for (int t = 0; t < 4; t++)
#pragma unroll
            for (int r = 0; r < 8; r++) acc[t].f[r] *= alpha;

        // repack P^T (C layout) -> B layout for the PV WMMA
        float o0[8], o1[8];
#pragma unroll
        for (int r = 0; r < 8; r++) { o0[r] = swap16f(p0[r]); o1[r] = swap16f(p1[r]); }
        float lo[8], hi[8];
#pragma unroll
        for (int r = 0; r < 8; r++) {
            lo[r] = half ? o1[r] : p0[r];   // rows 0..7  of tile 'half'
            hi[r] = half ? p1[r] : o0[r];   // rows 8..15 of tile 'half'
        }
        V16U pb;
        pb.q[0].x = pack2(lo[0], lo[1]);
        pb.q[0].y = pack2(lo[2], lo[3]);
        pb.q[0].z = pack2(lo[4], lo[5]);
        pb.q[0].w = pack2(lo[6], lo[7]);
        pb.q[1].x = pack2(hi[0], hi[1]);
        pb.q[1].y = pack2(hi[2], hi[3]);
        pb.q[1].z = pack2(hi[4], hi[5]);
        pb.q[1].w = pack2(hi[6], hi[7]);

#pragma unroll
        for (int t = 0; t < 4; t++) {
            const uint16_t* va = vT + (size_t)(16 * t + lane16) * 32;
            acc[t].v = WMMA_BF16(load_a16(va, 0, half), pb.v, acc[t].v);
        }
    }

    const float inv = 1.0f / l;
    uint16_t* obase = O + (tokBase + myq) * D_ + h * DK_;
#pragma unroll
    for (int t = 0; t < 4; t++) {
        uint4 w;
        w.x = pack2(acc[t].f[0] * inv, acc[t].f[1] * inv);
        w.y = pack2(acc[t].f[2] * inv, acc[t].f[3] * inv);
        w.z = pack2(acc[t].f[4] * inv, acc[t].f[5] * inv);
        w.w = pack2(acc[t].f[6] * inv, acc[t].f[7] * inv);
        *(uint4*)(obase + 16 * t + 8 * half) = w;
    }
}

// ---------------------------------------------------------------------------
// Host launch
// ---------------------------------------------------------------------------
extern "C" void kernel_launch(void* const* d_in, const int* in_sizes, int n_in,
                              void* d_out, int out_size, void* d_ws, size_t ws_size,
                              hipStream_t stream) {
    (void)in_sizes; (void)n_in; (void)out_size; (void)ws_size;
    const float* x  = (const float*)d_in[0];
    const int*   tp = (const int*)d_in[1];
    const float* Wq = (const float*)d_in[2];
    const float* Wk = (const float*)d_in[3];
    const float* Wv = (const float*)d_in[4];
    const float* Wo = (const float*)d_in[5];

    const size_t MB = 1ull << 20;
    uint8_t* ws = (uint8_t*)d_ws;
    uint16_t* xb  = (uint16_t*)(ws + 0 * MB);   // 8 MB  (reused later for O)
    uint16_t* wqb = (uint16_t*)(ws + 8 * MB);   // 2 MB
    uint16_t* wkb = (uint16_t*)(ws + 10 * MB);
    uint16_t* wvb = (uint16_t*)(ws + 12 * MB);
    uint16_t* wob = (uint16_t*)(ws + 14 * MB);
    uint16_t* Qb  = (uint16_t*)(ws + 16 * MB);  // 8 MB
    uint16_t* Kb  = (uint16_t*)(ws + 24 * MB);  // 8 MB
    uint16_t* Vb  = (uint16_t*)(ws + 32 * MB);  // 8 MB
    uint16_t* Vtb = (uint16_t*)(ws + 40 * MB);  // 8 MB
    uint16_t* Ob  = xb;                         // reuse (x dead after QKV)

    const int nX = NTOK * D_;   // 4,194,304
    const int nW = D_ * D_;     // 1,048,576

    cvt_f32_bf16<<<(nX + 255) / 256, 256, 0, stream>>>(x,  xb,  nX);
    cvt_f32_bf16<<<(nW + 255) / 256, 256, 0, stream>>>(Wq, wqb, nW);
    cvt_f32_bf16<<<(nW + 255) / 256, 256, 0, stream>>>(Wk, wkb, nW);
    cvt_f32_bf16<<<(nW + 255) / 256, 256, 0, stream>>>(Wv, wvb, nW);
    cvt_f32_bf16<<<(nW + 255) / 256, 256, 0, stream>>>(Wo, wob, nW);

    // projections: waves = (D/32) * (NTOK/64) = 32*64 = 2048 -> 256 blocks
    const int gemmBlocks = (D_ / 32) * (NTOK / 64) / 8;
    gemm_wt_xt<<<gemmBlocks, 256, 0, stream>>>(xb, wqb, Qb, D_, D_, D_ / 32, 0);
    gemm_wt_xt<<<gemmBlocks, 256, 0, stream>>>(xb, wkb, Kb, D_, D_, D_ / 32, 0);
    gemm_wt_xt<<<gemmBlocks, 256, 0, stream>>>(xb, wvb, Vb, D_, D_, D_ / 32, 0);

    const int ropeThreads = NTOK * (D_ / 2);   // 2,097,152
    rope_kernel<<<ropeThreads / 256, 256, 0, stream>>>(Qb, tp);
    rope_kernel<<<ropeThreads / 256, 256, 0, stream>>>(Kb, tp);

    transpose_v<<<nX / 256, 256, 0, stream>>>(Vb, Vtb);

    // attention: B*H*(S/128) = 512 blocks of 8 waves
    attn_kernel<<<B_ * H_ * (S_ / 128), 256, 0, stream>>>(Qb, Kb, Vtb, Ob);

    // output projection, f32 result
    gemm_wt_xt<<<gemmBlocks, 256, 0, stream>>>(Ob, wob, d_out, D_, D_, D_ / 32, 1);
}
